// AllDSVTBlocksTRT_8134668059008
// MI455X (gfx1250) — compile-verified, matrix operand order checked by reading, not run on hardware
//
#include <hip/hip_runtime.h>
#include <hip/hip_bf16.h>

// ---------------- problem constants ----------------
#define NN   24000
#define SS   700
#define KK   36
#define DD   192
#define HH   8
#define DHH  24
#define DFFN 384
#define LL   16
#define BB   8
#define SK   (SS * KK)        // 25200
#define EPSL 1e-5f

#define CDIV(a, b) (((a) + (b) - 1) / (b))

typedef __attribute__((ext_vector_type(16))) __bf16 v16bf;
typedef __attribute__((ext_vector_type(8)))  float  v8f;

union FragBF { uint4 u[2]; v16bf v; };

// ---------------- helpers ----------------
__device__ __forceinline__ float wave_reduce_sum(float v) {
    #pragma unroll
    for (int off = 16; off > 0; off >>= 1) v += __shfl_xor(v, off, 32);
    return v;
}

__device__ __forceinline__ float gelu_exact(float x) {
    return 0.5f * x * (1.f + erff(x * 0.70710678118654752f));
}

// ---------------- first-occurrence (segment_min) ----------------
__global__ void init_first_kernel(int* first, int total) {
    int t = blockIdx.x * blockDim.x + threadIdx.x;
    if (t < total) first[t] = 0x7fffffff;
}

__global__ void fill_first_kernel(const int* __restrict__ inds, int* __restrict__ first, int total) {
    int t = blockIdx.x * blockDim.x + threadIdx.x;
    if (t < total) atomicMin(&first[inds[t]], t);
}

// ---------------- weight packing into WMMA B-fragment order ----------------
// W: rowmajor (Nout x Kdim) fp32.  B[k][n] = W[n][k].
// Bpack layout: [ntile][kstep][lane][16 bf16]  (32 contiguous bytes per lane)
// lane: n = ntile*16 + (lane&15), h = lane>>4
// elem j: k = kstep*32 + (j<8 ? h*8+j : 16 + h*8 + (j-8))
__global__ void pack_weight_kernel(const float* __restrict__ W, __hip_bfloat16* __restrict__ Bp,
                                   int Nout, int Kdim) {
    int t = blockIdx.x * blockDim.x + threadIdx.x;
    int total = Nout * Kdim;
    if (t >= total) return;
    int j    = t & 15;
    int lane = (t >> 4) & 31;
    int rest = t >> 9;
    int Ksteps = Kdim >> 5;
    int ks = rest % Ksteps;
    int tn = rest / Ksteps;
    int n  = tn * 16 + (lane & 15);
    int h  = lane >> 4;
    int k  = ks * 32 + (j < 8 ? h * 8 + j : 16 + h * 8 + (j - 8));
    Bp[t] = __float2bfloat16(W[(size_t)n * Kdim + k]);
}

// ---------------- bf16 WMMA GEMM, 4-way N register blocking ----------------
// One wave computes a 16x64 output strip (4 WMMA accumulators, A fragment reused 4x).
// EPI==0: C[m][coloff+n] = acc + bias            (fp32 out)
// EPI==1: Cb[m][coloff+n] = bf16(gelu(acc+bias)) (bf16 out, fused GELU)
template <int EPI>
__global__ __launch_bounds__(256) void gemm_bf16_wmma(
    const __hip_bfloat16* __restrict__ A,      // M x Kdim, rowmajor bf16
    const __hip_bfloat16* __restrict__ Bp,     // packed fragments
    const float* __restrict__ bias,            // per output column (indexed with coloff)
    float* __restrict__ C,                     // M x ldc fp32 (EPI==0)
    __hip_bfloat16* __restrict__ Cb,           // M x ldc bf16 (EPI==1)
    int M, int Kdim, int Ntiles, int ntbase, int ldc, int coloff) {
    int wave = threadIdx.x >> 5;
    int lane = threadIdx.x & 31;
    int Mtiles = M >> 4;
    int Nt4 = Ntiles >> 2;                     // Ntiles is always a multiple of 4 here
    int t = blockIdx.x * 8 + wave;
    if (t >= Mtiles * Nt4) return;             // wave-uniform: EXEC stays all-ones for WMMA
    int tm  = t % Mtiles;
    int tn4 = t / Mtiles;
    int ln = lane & 15;
    int lh = lane >> 4;
    int Ksteps = Kdim >> 5;

    const __hip_bfloat16* Arow = A + (size_t)(tm * 16 + ln) * Kdim;
    const __hip_bfloat16* Bb[4];
    #pragma unroll
    for (int i = 0; i < 4; ++i)
        Bb[i] = Bp + ((size_t)(ntbase + tn4 * 4 + i) * Ksteps * 32 + lane) * 16;

    v8f acc[4];
    #pragma unroll
    for (int i = 0; i < 4; ++i) acc[i] = (v8f){0.f, 0.f, 0.f, 0.f, 0.f, 0.f, 0.f, 0.f};

    for (int ks = 0; ks < Ksteps; ++ks) {
        FragBF a;
        a.u[0] = *(const uint4*)(Arow + ks * 32 + lh * 8);        // elems 0..7  : K = ks*32 + lh*8 + j
        a.u[1] = *(const uint4*)(Arow + ks * 32 + 16 + lh * 8);   // elems 8..15 : K = ks*32 + 16 + lh*8 + j
        FragBF b[4];
        #pragma unroll
        for (int i = 0; i < 4; ++i) {
            const uint4* pb = (const uint4*)(Bb[i] + (size_t)ks * 512);
            b[i].u[0] = pb[0];
            b[i].u[1] = pb[1];
        }
        // four independent WMMAs (no RAW chain within a k-step)
        #pragma unroll
        for (int i = 0; i < 4; ++i)
            acc[i] = __builtin_amdgcn_wmma_f32_16x16x32_bf16(
                false, a.v, false, b[i].v, (short)0, acc[i], false, false);
    }

    #pragma unroll
    for (int i = 0; i < 4; ++i) {
        int ncol = coloff + (tn4 * 4 + i) * 16 + ln;
        float bv = bias[ncol];
        #pragma unroll
        for (int j = 0; j < 8; ++j) {
            int m = tm * 16 + lh * 8 + j;      // C/D layout: VGPR j -> M = j (+8 for lanes 16..31)
            float y = acc[i][j] + bv;
            if (EPI == 1) {
                Cb[(size_t)m * ldc + ncol] = __float2bfloat16(gelu_exact(y));
            } else {
                C[(size_t)m * ldc + ncol] = y;
            }
        }
    }
}

// ---------------- gather + positional add, fp32 -> bf16 ----------------
__global__ void gather_kernel(const float* __restrict__ cur, const float* __restrict__ pos,
                              const int* __restrict__ inds,
                              __hip_bfloat16* __restrict__ featb, __hip_bfloat16* __restrict__ qkb,
                              int total) {
    int t = blockIdx.x * blockDim.x + threadIdx.x;
    if (t >= total) return;
    int sk = t / DD;
    int d  = t - sk * DD;
    int v  = inds[sk];
    float f = cur[(size_t)v * DD + d];
    featb[t] = __float2bfloat16(f);
    qkb[t]   = __float2bfloat16(f + pos[(size_t)v * DD + d]);
}

// ---------------- per-set per-head attention (K=36, DH=24) ----------------
__global__ __launch_bounds__(64) void attn_kernel(const float* __restrict__ qkv,
                                                  const unsigned char* __restrict__ mask,
                                                  __hip_bfloat16* __restrict__ obf) {
    int s = blockIdx.x / HH;
    int h = blockIdx.x % HH;
    __shared__ float kt[KK][DHH];
    __shared__ float vt[KK][DHH];
    int tid = threadIdx.x;
    for (int i = tid; i < KK * DHH; i += 64) {
        int r = i / DHH, c = i - r * DHH;
        size_t base = ((size_t)(s * KK + r)) * 576;
        kt[r][c] = qkv[base + 192 + h * DHH + c];
        vt[r][c] = qkv[base + 384 + h * DHH + c];
    }
    __syncthreads();
    if (tid >= KK) return;

    float q[DHH];
    size_t qbase = ((size_t)(s * KK + tid)) * 576 + h * DHH;
    #pragma unroll
    for (int c = 0; c < DHH; ++c) q[c] = qkv[qbase + c];

    const unsigned char* mrow = mask + (size_t)s * KK;
    const float scale = 0.2041241452319315f;   // 1/sqrt(24)
    float sc[KK];
    float mx = -3.0e38f;
    for (int j = 0; j < KK; ++j) {
        float d;
        if (mrow[j]) {
            d = -1.0e30f;
        } else {
            d = 0.f;
            #pragma unroll
            for (int c = 0; c < DHH; ++c) d += q[c] * kt[j][c];
            d *= scale;
        }
        sc[j] = d;
        mx = fmaxf(mx, d);
    }
    float sum = 0.f;
    for (int j = 0; j < KK; ++j) { float p = __expf(sc[j] - mx); sc[j] = p; sum += p; }
    float inv = 1.f / sum;

    float o[DHH];
    #pragma unroll
    for (int c = 0; c < DHH; ++c) o[c] = 0.f;
    for (int j = 0; j < KK; ++j) {
        float w = sc[j] * inv;
        #pragma unroll
        for (int c = 0; c < DHH; ++c) o[c] += w * vt[j][c];
    }
    size_t obase = ((size_t)(s * KK + tid)) * DD + h * DHH;
    #pragma unroll
    for (int c = 0; c < DHH; ++c) obf[obase + c] = __float2bfloat16(o[c]);
}

// ---------------- scatter(first) + residual + LayerNorm, outputs fp32 + bf16 ----------------
__global__ __launch_bounds__(256) void scatter_ln_kernel(
    const float* __restrict__ cur, const float* __restrict__ oproj,
    const int* __restrict__ first, const float* __restrict__ g, const float* __restrict__ b,
    float* __restrict__ x, __hip_bfloat16* __restrict__ xb, int rows) {
    int wave = threadIdx.x >> 5;
    int lane = threadIdx.x & 31;
    int n = blockIdx.x * 8 + wave;
    if (n >= rows) return;
    int fi = first[n];
    const float* pa = cur + (size_t)n * DD;
    const float* po = oproj + (size_t)fi * DD;
    float vals[6];
    float s = 0.f;
    #pragma unroll
    for (int j = 0; j < 6; ++j) { int d = lane + j * 32; vals[j] = pa[d] + po[d]; s += vals[j]; }
    s = wave_reduce_sum(s);
    float mean = s * (1.f / DD);
    float vs = 0.f;
    #pragma unroll
    for (int j = 0; j < 6; ++j) { float dd = vals[j] - mean; vs += dd * dd; }
    vs = wave_reduce_sum(vs);
    float rstd = rsqrtf(vs * (1.f / DD) + EPSL);
    #pragma unroll
    for (int j = 0; j < 6; ++j) {
        int d = lane + j * 32;
        float y = (vals[j] - mean) * rstd * g[d] + b[d];
        x[(size_t)n * DD + d]  = y;
        xb[(size_t)n * DD + d] = __float2bfloat16(y);
    }
}

// ---------------- (a+b) then LayerNorm, fp32 out ----------------
__global__ __launch_bounds__(256) void add_ln_kernel(
    const float* __restrict__ a, const float* __restrict__ bsrc,
    const float* __restrict__ g, const float* __restrict__ be,
    float* __restrict__ out, int rows) {
    int wave = threadIdx.x >> 5;
    int lane = threadIdx.x & 31;
    int n = blockIdx.x * 8 + wave;
    if (n >= rows) return;
    const float* pa = a + (size_t)n * DD;
    const float* pb = bsrc + (size_t)n * DD;
    float vals[6];
    float s = 0.f;
    #pragma unroll
    for (int j = 0; j < 6; ++j) { int d = lane + j * 32; vals[j] = pa[d] + pb[d]; s += vals[j]; }
    s = wave_reduce_sum(s);
    float mean = s * (1.f / DD);
    float vs = 0.f;
    #pragma unroll
    for (int j = 0; j < 6; ++j) { float dd = vals[j] - mean; vs += dd * dd; }
    vs = wave_reduce_sum(vs);
    float rstd = rsqrtf(vs * (1.f / DD) + EPSL);
    #pragma unroll
    for (int j = 0; j < 6; ++j) {
        int d = lane + j * 32;
        out[(size_t)n * DD + d] = (vals[j] - mean) * rstd * g[d] + be[d];
    }
}

// ---------------- host orchestration ----------------
extern "C" void kernel_launch(void* const* d_in, const int* in_sizes, int n_in,
                              void* d_out, int out_size, void* d_ws, size_t ws_size,
                              hipStream_t stream) {
    (void)in_sizes; (void)n_in; (void)out_size; (void)ws_size;

    const float*         pillar = (const float*)d_in[0];
    const int*           inds0  = (const int*)d_in[1];            // (2,S,K)
    const int*           inds1  = (const int*)d_in[2];
    const unsigned char* mask0  = (const unsigned char*)d_in[3];  // (2,S,K) bool
    const unsigned char* mask1  = (const unsigned char*)d_in[4];
    const float*         pos    = (const float*)d_in[5];          // (B,2,N,D)
    const float*         Wqkv   = (const float*)d_in[6];          // (L,576,192)
    const float*         bqkv   = (const float*)d_in[7];          // (L,576)
    const float*         Wo     = (const float*)d_in[8];          // (L,192,192)
    const float*         bo     = (const float*)d_in[9];
    const float*         ln1g   = (const float*)d_in[10];
    const float*         ln1b   = (const float*)d_in[11];
    const float*         W1     = (const float*)d_in[12];         // (L,384,192)
    const float*         b1     = (const float*)d_in[13];
    const float*         W2     = (const float*)d_in[14];         // (L,192,384)
    const float*         b2     = (const float*)d_in[15];
    const float*         ln2g   = (const float*)d_in[16];
    const float*         ln2b   = (const float*)d_in[17];
    const float*         flng   = (const float*)d_in[18];         // (B,D)
    const float*         flnb   = (const float*)d_in[19];

    // ---- workspace layout (aliased across sequential phases) ----
    char* ws = (char*)d_ws;
    size_t off = 0;
    auto alloc = [&](size_t sz) { size_t r = off; off += (sz + 255) & ~(size_t)255; return r; };
    const size_t NB = (size_t)NN * DD * 4;

    float*          cur     = (float*)(ws + alloc(NB));
    float*          resid   = (float*)(ws + alloc(NB));
    float*          xbuf    = (float*)(ws + alloc(NB));
    int*            firstp  = (int*)  (ws + alloc((size_t)4 * NN * 4));
    __hip_bfloat16* packQKV = (__hip_bfloat16*)(ws + alloc((size_t)576 * 192 * 2));
    __hip_bfloat16* packWo  = (__hip_bfloat16*)(ws + alloc((size_t)192 * 192 * 2));
    __hip_bfloat16* packW1  = (__hip_bfloat16*)(ws + alloc((size_t)384 * 192 * 2));
    __hip_bfloat16* packW2  = (__hip_bfloat16*)(ws + alloc((size_t)192 * 384 * 2));
    char*           bigA    = ws + alloc((size_t)SK * 576 * 4);     // qkv -> oproj
    char*           regF    = ws + alloc((size_t)NN * DFFN * 2);    // featb -> h1b
    char*           regG    = ws + alloc(NB);                       // qkb  -> ff
    char*           regH    = ws + alloc((size_t)SK * DD * 2);      // obf  -> xb

    float*          qkvbuf = (float*)bigA;
    float*          oproj  = (float*)bigA;
    __hip_bfloat16* featb  = (__hip_bfloat16*)regF;
    __hip_bfloat16* h1b    = (__hip_bfloat16*)regF;
    __hip_bfloat16* qkb    = (__hip_bfloat16*)regG;
    float*          ff     = (float*)regG;
    __hip_bfloat16* obf    = (__hip_bfloat16*)regH;
    __hip_bfloat16* xb     = (__hip_bfloat16*)regH;

    // ---- init ----
    hipMemcpyAsync(cur, pillar, NB, hipMemcpyDeviceToDevice, stream);

    init_first_kernel<<<CDIV(4 * NN, 256), 256, 0, stream>>>(firstp, 4 * NN);
    fill_first_kernel<<<CDIV(SK, 256), 256, 0, stream>>>(inds0,      firstp + 0 * NN, SK);
    fill_first_kernel<<<CDIV(SK, 256), 256, 0, stream>>>(inds0 + SK, firstp + 1 * NN, SK);
    fill_first_kernel<<<CDIV(SK, 256), 256, 0, stream>>>(inds1,      firstp + 2 * NN, SK);
    fill_first_kernel<<<CDIV(SK, 256), 256, 0, stream>>>(inds1 + SK, firstp + 3 * NN, SK);

    const int MtSK = SK / 16;   // 1575
    const int MtN  = NN / 16;   // 1500

    for (int blc = 0; blc < BB; ++blc) {
        hipMemcpyAsync(resid, cur, NB, hipMemcpyDeviceToDevice, stream);
        int shift = blc & 1;
        for (int set = 0; set < 2; ++set) {
            int li = blc * 2 + set;
            int gi = shift * 2 + set;
            const int*           inds = (shift ? inds1 : inds0) + (size_t)set * SK;
            const unsigned char* msk  = (shift ? mask1 : mask0) + (size_t)set * SK;
            const float*         posl = pos + ((size_t)blc * 2 + set) * (size_t)NN * DD;

            // pack this layer's weights into WMMA B-fragment order
            pack_weight_kernel<<<CDIV(576 * 192, 256), 256, 0, stream>>>(
                Wqkv + (size_t)li * 576 * 192, packQKV, 576, 192);
            pack_weight_kernel<<<CDIV(192 * 192, 256), 256, 0, stream>>>(
                Wo + (size_t)li * 192 * 192, packWo, 192, 192);
            pack_weight_kernel<<<CDIV(384 * 192, 256), 256, 0, stream>>>(
                W1 + (size_t)li * 384 * 192, packW1, 384, 192);
            pack_weight_kernel<<<CDIV(192 * 384, 256), 256, 0, stream>>>(
                W2 + (size_t)li * 192 * 384, packW2, 192, 384);

            // gather + pos
            gather_kernel<<<CDIV(SK * DD, 256), 256, 0, stream>>>(
                cur, posl, inds, featb, qkb, SK * DD);

            // Q,K projection (cols 0..383) from qk ; V (cols 384..575) from feat
            gemm_bf16_wmma<0><<<CDIV(MtSK * 6, 8), 256, 0, stream>>>(
                qkb, packQKV, bqkv + (size_t)li * 576, qkvbuf, nullptr,
                SK, 192, 24, 0, 576, 0);
            gemm_bf16_wmma<0><<<CDIV(MtSK * 3, 8), 256, 0, stream>>>(
                featb, packQKV, bqkv + (size_t)li * 576, qkvbuf, nullptr,
                SK, 192, 12, 24, 576, 384);

            // attention per (set, head)
            attn_kernel<<<SS * HH, 64, 0, stream>>>(qkvbuf, msk, obf);

            // output projection
            gemm_bf16_wmma<0><<<CDIV(MtSK * 3, 8), 256, 0, stream>>>(
                obf, packWo, bo + (size_t)li * 192, oproj, nullptr,
                SK, 192, 12, 0, 192, 0);

            // scatter first-occurrence + residual + LN1
            scatter_ln_kernel<<<NN / 8, 256, 0, stream>>>(
                cur, oproj, firstp + (size_t)gi * NN,
                ln1g + (size_t)li * DD, ln1b + (size_t)li * DD, xbuf, xb, NN);

            // FFN: GEMM1 with fused bias+GELU -> bf16, then GEMM2
            gemm_bf16_wmma<1><<<CDIV(MtN * 6, 8), 256, 0, stream>>>(
                xb, packW1, b1 + (size_t)li * DFFN, nullptr, h1b,
                NN, 192, 24, 0, 384, 0);
            gemm_bf16_wmma<0><<<CDIV(MtN * 3, 8), 256, 0, stream>>>(
                h1b, packW2, b2 + (size_t)li * DD, ff, nullptr,
                NN, 384, 12, 0, 192, 0);

            // residual + LN2 -> cur
            add_ln_kernel<<<NN / 8, 256, 0, stream>>>(
                xbuf, ff, ln2g + (size_t)li * DD, ln2b + (size_t)li * DD, cur, NN);
        }
        // block-level residual + final LN -> cur
        add_ln_kernel<<<NN / 8, 256, 0, stream>>>(
            resid, cur, flng + (size_t)blc * DD, flnb + (size_t)blc * DD, cur, NN);
    }

    hipMemcpyAsync(d_out, cur, NB, hipMemcpyDeviceToDevice, stream);
}
